// MaskDINOLateFusion_74620761801513
// MI455X (gfx1250) — matrix operand order repaired and checked by reading, not compiled, not run on hardware
//
#include <hip/hip_runtime.h>
#include <math.h>

#define HID 256
#define NHEAD 8
#define DH 32
#define NPOINT 4
#define NLVL 4
#define NLAYER 9
#define NQ 300
#define NTXT 80
#define DFF 2048
#define BS 2
#define LSRC 12240
#define LTOT 380
#define MHW (192*192)

typedef __attribute__((ext_vector_type(16))) _Float16 v16h;
typedef __attribute__((ext_vector_type(8)))  float    v8f;

// ---------------------------------------------------------------------------
// f32 -> f16 operand conversion (run once per B operand; weights/txt/mfeat)
// ---------------------------------------------------------------------------
__global__ void f32_to_f16_k(const float* __restrict__ in, _Float16* __restrict__ out,
                             long long n)
{
  long long i = (long long)blockIdx.x * 256 + threadIdx.x;
  if (i < n) out[i] = (_Float16)in[i];
}

// ---------------------------------------------------------------------------
// WMMA GEMM:  C[M,N] = A[M,K] x op(B) + bias
//   A: fp32 (converted to f16 in-register, amortized over 4 N-tiles)
//   B: pre-converted f16 (fragments load directly as v16h -> 2x b128, no cvt)
//   BT=1 : B is [N,K] row-major;  BT=0 : B is [K,N] row-major
// One wave computes a 16x64 strip of C (4 WMMA tiles, A fragment reused).
// Branchless operand fetch: indices clamped, never predicated.
// Fragment layouts per CDNA5 ISA §7.12.2 (wave32).
// ---------------------------------------------------------------------------
__device__ __forceinline__ void pack8(v16h& v, int base, float4 p, float4 q)
{
  v[base + 0] = (_Float16)p.x; v[base + 1] = (_Float16)p.y;
  v[base + 2] = (_Float16)p.z; v[base + 3] = (_Float16)p.w;
  v[base + 4] = (_Float16)q.x; v[base + 5] = (_Float16)q.y;
  v[base + 6] = (_Float16)q.z; v[base + 7] = (_Float16)q.w;
}

template<int BT, int RELU>
__global__ __launch_bounds__(32) void wmma_gemm(
    const float* __restrict__ A, const _Float16* __restrict__ B,
    const float* __restrict__ bias, float* __restrict__ C,
    int M, int N, int K, long long sA, long long sB, long long sC)
{
  int bz = blockIdx.z;
  A += (long long)bz * sA;
  B += (long long)bz * sB;
  C += (long long)bz * sC;
  int n0 = blockIdx.x * 64;
  int m0 = blockIdx.y * 16;
  int lane = threadIdx.x & 31;
  int half = lane >> 4;
  int r    = lane & 15;
  int amc  = min(m0 + r, M - 1);           // clamped A row (branchless loads)
  int bnc[4];
#pragma unroll
  for (int t = 0; t < 4; ++t) bnc[t] = min(n0 + t * 16 + r, N - 1);

  v8f acc[4] = {};
  for (int k0 = 0; k0 < K; k0 += 32) {
    // A 16x32 f16 fragment: lane row amc, K runs [k0+16h..+15]
    const float* abase = A + (long long)amc * K + k0 + 16 * half;
    float4 a0 = ((const float4*)abase)[0];
    float4 a1 = ((const float4*)abase)[1];
    float4 a2 = ((const float4*)(abase + 8))[0];
    float4 a3 = ((const float4*)(abase + 8))[1];
    // B fragments: pure f16 data, no conversion -> loads can all be in flight
    v16h bvs[4];
    if (BT) {
#pragma unroll
      for (int t = 0; t < 4; ++t)
        bvs[t] = *(const v16h*)(B + (long long)bnc[t] * K + k0 + 16 * half);
    } else {
#pragma unroll
      for (int t = 0; t < 4; ++t) {
        const _Float16* bbase = B + (long long)(k0 + 16 * half) * N + bnc[t];
#pragma unroll
        for (int e = 0; e < 16; ++e)
          bvs[t][e] = bbase[(long long)e * N];
      }
    }
    v16h av;
    pack8(av, 0, a0, a1);
    pack8(av, 8, a2, a3);
#pragma unroll
    for (int t = 0; t < 4; ++t)
      acc[t] = __builtin_amdgcn_wmma_f32_16x16x32_f16(
          false, av, false, bvs[t], (short)0, acc[t], false, false);
  }
#pragma unroll
  for (int t = 0; t < 4; ++t) {
    int n = n0 + t * 16 + r;
    if (n < N) {
      float bb = bias ? bias[n] : 0.f;
#pragma unroll
      for (int j = 0; j < 8; ++j) {
        int m = m0 + j + 8 * half;   // C: VGPR j -> row j (lanes 0-15) / 8+j
        if (m < M) {
          float v = acc[t][j] + bb;
          if (RELU) v = v > 0.f ? v : 0.f;
          C[(long long)m * N + n] = v;
        }
      }
    }
  }
}

static void gemm(bool bt, bool relu,
                 const float* A, const _Float16* B, const float* bias, float* C,
                 int M, int N, int K, long long sA, long long sB, long long sC,
                 int bz, hipStream_t st)
{
  dim3 gr((N + 63) / 64, (M + 15) / 16, bz), bl(32);
  if (bt) {
    if (relu) wmma_gemm<1,1><<<gr,bl,0,st>>>(A,B,bias,C,M,N,K,sA,sB,sC);
    else      wmma_gemm<1,0><<<gr,bl,0,st>>>(A,B,bias,C,M,N,K,sA,sB,sC);
  } else {
    if (relu) wmma_gemm<0,1><<<gr,bl,0,st>>>(A,B,bias,C,M,N,K,sA,sB,sC);
    else      wmma_gemm<0,0><<<gr,bl,0,st>>>(A,B,bias,C,M,N,K,sA,sB,sC);
  }
}

// ---------------------------------------------------------------------------
// Elementwise / reduction helpers
// ---------------------------------------------------------------------------
__device__ __forceinline__ float inv_sig(float x) {
  x = fminf(fmaxf(x, 0.f), 1.f);
  return logf(fmaxf(x, 1e-5f)) - logf(fmaxf(1.f - x, 1e-5f));
}
__device__ __forceinline__ float sigm(float x) { return 1.f / (1.f + __expf(-x)); }

__global__ void ew_add_k(const float* __restrict__ a, const float* __restrict__ b,
                         float* __restrict__ o, long long n)
{
  long long i = (long long)blockIdx.x * 256 + threadIdx.x;
  if (i < n) o[i] = a[i] + b[i];
}

// row layer-norm over C=256, optional residual, one wave per row
__global__ __launch_bounds__(32) void add_ln_k(
    const float* __restrict__ x, const float* __restrict__ res,
    const float* __restrict__ g, const float* __restrict__ be,
    float* __restrict__ out)
{
  long long row = blockIdx.x;
  int lane = threadIdx.x;
  const float* xr = x + row * HID;
  const float* rr = res ? res + row * HID : nullptr;
  float v[8]; float s = 0.f;
#pragma unroll
  for (int i = 0; i < 8; ++i) {
    float t = xr[lane * 8 + i] + (rr ? rr[lane * 8 + i] : 0.f);
    v[i] = t; s += t;
  }
  for (int m = 16; m > 0; m >>= 1) s += __shfl_xor(s, m, 32);
  float mean = s * (1.f / 256.f);
  float vs = 0.f;
#pragma unroll
  for (int i = 0; i < 8; ++i) { float d = v[i] - mean; vs += d * d; }
  for (int m = 16; m > 0; m >>= 1) vs += __shfl_xor(vs, m, 32);
  float inv = rsqrtf(vs * (1.f / 256.f) + 1e-5f);
#pragma unroll
  for (int i = 0; i < 8; ++i) {
    int c = lane * 8 + i;
    out[row * HID + c] = (v[i] - mean) * inv * g[c] + be[c];
  }
}

// ---------------------------------------------------------------------------
// Encoder-side prep
// ---------------------------------------------------------------------------
__global__ void build_src_k(const float* __restrict__ x0, const float* __restrict__ x1,
                            const float* __restrict__ x2, const float* __restrict__ x3,
                            float* __restrict__ src)
{
  long long i = (long long)blockIdx.x * 256 + threadIdx.x;
  long long n = (long long)BS * LSRC * HID;
  if (i >= n) return;
  int c = i & 255; long long t = i >> 8;
  int s = (int)(t % LSRC); int b = (int)(t / LSRC);
  const float* xp; int hw, loc;
  if (s < 144)       { xp = x3; hw = 144;  loc = s; }
  else if (s < 720)  { xp = x2; hw = 576;  loc = s - 144; }
  else if (s < 3024) { xp = x1; hw = 2304; loc = s - 720; }
  else               { xp = x0; hw = 9216; loc = s - 3024; }
  src[i] = xp[((long long)b * HID + c) * hw + loc];
}

__global__ void props_k(float* __restrict__ pu, float* __restrict__ valid)
{
  int s = blockIdx.x * 256 + threadIdx.x;
  if (s >= LSRC) return;
  int H, loc, lvl;
  if (s < 144)       { lvl = 0; H = 12; loc = s; }
  else if (s < 720)  { lvl = 1; H = 24; loc = s - 144; }
  else if (s < 3024) { lvl = 2; H = 48; loc = s - 720; }
  else               { lvl = 3; H = 96; loc = s - 3024; }
  int W = H, r = loc / W, c = loc % W;
  float p[4];
  p[0] = (c + 0.5f) / W; p[1] = (r + 0.5f) / H;
  p[2] = 0.05f * (float)(1 << lvl); p[3] = p[2];
  bool ok = true;
#pragma unroll
  for (int j = 0; j < 4; ++j) ok = ok && (p[j] > 0.01f) && (p[j] < 0.99f);
  valid[s] = ok ? 1.f : 0.f;
#pragma unroll
  for (int j = 0; j < 4; ++j)
    pu[s * 4 + j] = ok ? logf(p[j] / (1.f - p[j])) : __builtin_inff();
}

__global__ void mask_mem_k(const float* __restrict__ src, const float* __restrict__ valid,
                           float* __restrict__ o)
{
  long long i = (long long)blockIdx.x * 256 + threadIdx.x;
  long long n = (long long)BS * LSRC * HID;
  if (i >= n) return;
  int s = (int)((i >> 8) % LSRC);
  o[i] = (valid[s] != 0.f) ? src[i] : 0.f;
}

__global__ void coords_add_k(float* __restrict__ cu, const float* __restrict__ pu)
{
  long long i = (long long)blockIdx.x * 256 + threadIdx.x;
  long long n = (long long)BS * LSRC * 4;
  if (i >= n) return;
  cu[i] += pu[i % ((long long)LSRC * 4)];
}

__global__ void rowmax_k(const float* __restrict__ sc, float* __restrict__ o, long long rows)
{
  long long i = (long long)blockIdx.x * 256 + threadIdx.x;
  if (i >= rows) return;
  const float* p = sc + i * NTXT;
  float m = p[0];
  for (int j = 1; j < NTXT; ++j) m = fmaxf(m, p[j]);
  o[i] = m;
}

// deterministic top-k by rank counting (ties broken by lower index, like top_k)
__global__ void topk_rank_k(const float* __restrict__ smax, int* __restrict__ idx)
{
  int b = blockIdx.y;
  int i = blockIdx.x * 256 + threadIdx.x;
  if (i >= LSRC) return;
  const float* p = smax + (long long)b * LSRC;
  float v = p[i];
  int r = 0;
  for (int j = 0; j < LSRC; ++j) {
    float u = p[j];
    r += (u > v) || (u == v && j < i);
  }
  if (r < NQ) idx[b * NQ + r] = i;
}

__global__ void init_tgt_k(const float* __restrict__ mem, const int* __restrict__ topidx,
                           const float* __restrict__ lab, const float* __restrict__ pb,
                           const float* __restrict__ txt, float* __restrict__ tgt)
{
  long long i = (long long)blockIdx.x * 256 + threadIdx.x;
  long long n = (long long)BS * LTOT * HID;
  if (i >= n) return;
  int c = i & 255; long long t = i >> 8;
  int r = (int)(t % LTOT); int b = (int)(t / LTOT);
  float v;
  if (r < NQ) {
    int id = topidx[b * NQ + r];
    v = mem[((long long)b * LSRC + id) * HID + c];
  } else {
    v = lab[c] + pb[HID + c] + txt[((long long)b * NTXT + (r - NQ)) * HID + c];
  }
  tgt[i] = v;
}

__global__ void init_ref_k(const float* __restrict__ cu, const int* __restrict__ topidx,
                           float* __restrict__ ref)
{
  long long i = (long long)blockIdx.x * 256 + threadIdx.x;
  long long n = (long long)BS * LTOT * 4;
  if (i >= n) return;
  int j = i & 3; long long t = i >> 2;
  int r = (int)(t % LTOT); int b = (int)(t / LTOT);
  float u;
  if (r < NQ) {
    int id = topidx[b * NQ + r];
    u = cu[((long long)b * LSRC + id) * 4 + j];
  } else {
    u = (j < 2) ? logf(1e-5f) : -logf(1e-5f);   // inverse_sigmoid([0,0,1,1])
  }
  ref[i] = sigm(u);
}

__global__ void build_mask_k(float* __restrict__ m)
{
  int i = blockIdx.x * 256 + threadIdx.x;
  if (i >= LTOT * LTOT) return;
  int r = i / LTOT, c = i % LTOT;
  bool keep = (r < NQ && c < NQ) || (r >= NQ && c >= NQ && r == c);
  m[i] = keep ? 0.f : 1.f;
}

// ---------------------------------------------------------------------------
// Decoder pieces
// ---------------------------------------------------------------------------
__global__ void sine_embed_k(const float* __restrict__ ref, float* __restrict__ se)
{
  long long i = (long long)blockIdx.x * 256 + threadIdx.x;
  long long n = (long long)BS * LTOT * 512;
  if (i >= n) return;
  int c = i & 511; long long t = i >> 9;
  int q = (int)(t % LTOT); int b = (int)(t / LTOT);
  int g = c >> 7, j = c & 127;
  const int comp[4] = {1, 0, 2, 3};
  float v = ref[((long long)b * LTOT + q) * 4 + comp[g]];
  float dimt = powf(10000.f, (float)(2 * (j / 2)) / 128.f);
  float p = v * 6.283185307179586f / dimt;
  se[i] = (j & 1) ? cosf(p) : sinf(p);
}

// self-attention: one block per (query i, head h, batch b)
__global__ __launch_bounds__(128) void mha_attn_k(
    const float* __restrict__ qh, const float* __restrict__ kh,
    const float* __restrict__ vh, const float* __restrict__ mask,
    float* __restrict__ out)
{
  int i = blockIdx.x, h = blockIdx.y, b = blockIdx.z;
  __shared__ float s[LTOT];
  __shared__ float red[128];
  int t = threadIdx.x;
  const float scale = 0.17677669529663687f;   // 1/sqrt(32)
  const float* qrow = qh + ((long long)b * LTOT + i) * HID + h * DH;
  for (int j = t; j < LTOT; j += 128) {
    const float* krow = kh + ((long long)b * LTOT + j) * HID + h * DH;
    float d = 0.f;
#pragma unroll
    for (int c = 0; c < DH; ++c) d += qrow[c] * krow[c];
    d *= scale;
    if (mask[i * LTOT + j] != 0.f) d = -__builtin_inff();
    s[j] = d;
  }
  __syncthreads();
  float m = -__builtin_inff();
  for (int j = t; j < LTOT; j += 128) m = fmaxf(m, s[j]);
  red[t] = m; __syncthreads();
  for (int k = 64; k > 0; k >>= 1) { if (t < k) red[t] = fmaxf(red[t], red[t + k]); __syncthreads(); }
  m = red[0]; __syncthreads();
  float sum = 0.f;
  for (int j = t; j < LTOT; j += 128) { float e = __expf(s[j] - m); s[j] = e; sum += e; }
  red[t] = sum; __syncthreads();
  for (int k = 64; k > 0; k >>= 1) { if (t < k) red[t] += red[t + k]; __syncthreads(); }
  float inv = 1.f / red[0]; __syncthreads();
  if (t < DH) {
    float acc = 0.f;
    for (int j = 0; j < LTOT; ++j)
      acc += s[j] * vh[((long long)b * LTOT + j) * HID + h * DH + t];
    out[((long long)b * LTOT + i) * HID + h * DH + t] = acc * inv;
  }
}

__global__ void softmax16_k(float* __restrict__ aw, long long n)   // n = BS*LTOT*NHEAD
{
  long long i = (long long)blockIdx.x * 256 + threadIdx.x;
  if (i >= n) return;
  float* p = aw + i * 16;
  float m = p[0];
#pragma unroll
  for (int j = 1; j < 16; ++j) m = fmaxf(m, p[j]);
  float s = 0.f;
#pragma unroll
  for (int j = 0; j < 16; ++j) { float e = __expf(p[j] - m); p[j] = e; s += e; }
  float inv = 1.f / s;
#pragma unroll
  for (int j = 0; j < 16; ++j) p[j] *= inv;
}

// deformable attention sampling: one wave per (q,h,b), lane = channel d
__global__ __launch_bounds__(32) void deform_attn_k(
    const float* __restrict__ val, const float* __restrict__ ref,
    const float* __restrict__ off, const float* __restrict__ aw,
    float* __restrict__ out)
{
  int q = blockIdx.x, h = blockIdx.y, b = blockIdx.z;
  int d = threadIdx.x;
  const int HH[4] = {12, 24, 48, 96};
  const int ST[4] = {0, 144, 720, 3024};
  const float* rp = ref + ((long long)b * LTOT + q) * 4;
  float rx = rp[0], ry = rp[1], rw = rp[2], rh = rp[3];
  const float* ob = off + ((long long)b * LTOT + q) * HID + h * 32;   // [lvl*8+pt*2+{0,1}]
  const float* ab = aw  + ((long long)b * LTOT + q) * 128 + h * 16;   // [lvl*4+pt]
  float acc = 0.f;
  for (int lvl = 0; lvl < 4; ++lvl) {
    int H = HH[lvl], W = H, st = ST[lvl];
    for (int pt = 0; pt < 4; ++pt) {
      float ox = ob[lvl * 8 + pt * 2 + 0];
      float oy = ob[lvl * 8 + pt * 2 + 1];
      float lx = (rx + ox * 0.25f * rw * 0.5f) * W - 0.5f;
      float ly = (ry + oy * 0.25f * rh * 0.5f) * H - 0.5f;
      float x0 = floorf(lx), y0 = floorf(ly);
      float fx = lx - x0, fy = ly - y0;
      float a = ab[lvl * 4 + pt];
      float sv = 0.f;
#pragma unroll
      for (int dy = 0; dy < 2; ++dy)
#pragma unroll
        for (int dx = 0; dx < 2; ++dx) {
          float xi = x0 + dx, yi = y0 + dy;
          float w = (dx ? fx : 1.f - fx) * (dy ? fy : 1.f - fy);
          bool ok = (xi >= 0.f) && (xi < (float)W) && (yi >= 0.f) && (yi < (float)H);
          int xc = (int)fminf(fmaxf(xi, 0.f), (float)(W - 1));
          int yc = (int)fminf(fmaxf(yi, 0.f), (float)(H - 1));
          float gv = val[((long long)b * LSRC + (st + yc * W + xc)) * HID + h * 32 + d];
          sv += (ok ? w : 0.f) * gv;
        }
      acc += a * sv;
    }
  }
  out[((long long)b * LTOT + q) * HID + h * 32 + d] = acc;
}

__global__ void bbox_sigmoid_k(const float* __restrict__ bb, const float* __restrict__ ref,
                               float* __restrict__ out, long long n)
{
  long long i = (long long)blockIdx.x * 256 + threadIdx.x;
  if (i >= n) return;
  out[i] = sigm(bb[i] + inv_sig(ref[i]));
}

// ---------------------------------------------------------------------------
// Orchestration
// ---------------------------------------------------------------------------
extern "C" void kernel_launch(void* const* d_in, const int* in_sizes, int n_in,
                              void* d_out, int out_size, void* d_ws, size_t ws_size,
                              hipStream_t stream)
{
  (void)in_sizes; (void)n_in; (void)out_size; (void)ws_size;
  auto F = [&](int i) -> const float* { return (const float*)d_in[i]; };
  const float *x0 = F(0), *x1 = F(1), *x2 = F(2), *x3 = F(3), *mfeat = F(4), *txt = F(5);
  // params flattened (pytree: dict keys sorted)
  const float *bb1w=F(6), *bb1b=F(7), *bb2w=F(8), *bb2b=F(9), *bb3w=F(10), *bb3b=F(11);
  const float *clsw=F(12), *decb=F(13), *decg=F(14), *encbias=F(15);
  const float *eb1w=F(16), *eb1b=F(17), *eb2w=F(18), *eb2b=F(19), *eb3w=F(20), *eb3b=F(21);
  const float *enclnb=F(22), *enclng=F(23), *encw=F(24), *lab=F(25);
  const float *me1w=F(224), *me1b=F(225), *me2w=F(226), *me2b=F(227), *me3w=F(228), *me3b=F(229);
  const float *pb=F(230), *rp1w=F(231), *rp1b=F(232), *rp2w=F(233), *rp2b=F(234);

  size_t off = 0;
  auto allocb = [&](size_t bytes) -> void* {
    void* p = (void*)((char*)d_ws + off);
    off += (bytes + 255) & ~(size_t)255;
    return p;
  };
  auto alloc  = [&](size_t n) -> float*    { return (float*)allocb(n * 4); };
  auto alloch = [&](size_t n) -> _Float16* { return (_Float16*)allocb(n * 2); };

  auto g1 = [](long long n) { return dim3((unsigned)((n + 255) / 256)); };
  // convert an f32 tensor to a fresh f16 buffer
  auto cvt = [&](const float* src, size_t n) -> _Float16* {
    _Float16* d = alloch(n);
    f32_to_f16_k<<<g1((long long)n), 256, 0, stream>>>(src, d, (long long)n);
    return d;
  };

  const size_t SB = (size_t)BS * LSRC * HID;   // 6.27M floats
  const size_t DB = (size_t)BS * LTOT * HID;   // 194,560 floats
  float* SRC  = alloc(SB);
  float* MEM  = alloc(SB);
  float* T1   = alloc(SB);
  float* T2   = alloc(SB);
  float* SCR  = alloc((size_t)BS * LSRC * NTXT);
  float* CU   = alloc((size_t)BS * LSRC * 4);
  float* PU   = alloc((size_t)LSRC * 4);
  float* VAL  = alloc(LSRC);
  float* SMX  = alloc((size_t)BS * LSRC);
  int*   TIDX = (int*)alloc(BS * NQ);
  float* OUT  = alloc(DB);
  float* QPOS = alloc(DB);
  float* Q    = alloc(DB);
  float* SE   = alloc((size_t)BS * LTOT * 512);
  float* QH   = alloc(DB);
  float* KH   = alloc(DB);
  float* VH   = alloc(DB);
  float* ATT  = alloc(DB);
  float* DT1  = alloc(DB);
  float* DT2  = alloc(DB);
  float* FFB  = alloc((size_t)BS * LTOT * DFF);
  float* OFFB = alloc(DB);
  float* AW   = alloc((size_t)BS * LTOT * 128);
  float* BB   = alloc((size_t)BS * LTOT * 4);
  float* REF  = alloc((size_t)BS * LTOT * 4);
  float* MASK = alloc((size_t)LTOT * LTOT);
  float* HLST = alloc(DB);
  float* DN   = alloc(DB);
  float* CW   = alloc(DB);
  float* ME1  = alloc((size_t)BS * NQ * HID);
  float* ME2  = alloc((size_t)BS * NQ * HID);
  float* MEB  = alloc((size_t)BS * NQ * HID);

  // ---- one-time f16 conversions of every GEMM B-operand ----
  _Float16* encwh = cvt(encw, 256*256);
  _Float16* clswh = cvt(clsw, 256*256);          // used in BN ([K,N]) mode
  _Float16* eb1wh = cvt(eb1w, 256*256);
  _Float16* eb2wh = cvt(eb2w, 256*256);
  _Float16* eb3wh = cvt(eb3w, 4*256);
  _Float16* rp1wh = cvt(rp1w, 256*512);
  _Float16* rp2wh = cvt(rp2w, 256*256);
  _Float16* bb1wh = cvt(bb1w, 256*256);
  _Float16* bb2wh = cvt(bb2w, 256*256);
  _Float16* bb3wh = cvt(bb3w, 4*256);
  _Float16* me1wh = cvt(me1w, 256*256);
  _Float16* me2wh = cvt(me2w, 256*256);
  _Float16* me3wh = cvt(me3w, 256*256);
  _Float16* txth  = cvt(txt,  (size_t)BS*NTXT*HID);
  _Float16* mfh   = cvt(mfeat,(size_t)BS*HID*MHW);   // BN mode B
  _Float16 *awwh[NLAYER], *owh[NLAYER], *offwh[NLAYER], *vwh[NLAYER];
  _Float16 *f1wh[NLAYER], *f2wh[NLAYER], *inwh[NLAYER], *outwh[NLAYER];
  for (int l = 0; l < NLAYER; ++l) {
    int p0 = 26 + 22 * l;
    awwh[l]  = cvt(F(p0+1),  128*256);
    owh[l]   = cvt(F(p0+3),  256*256);
    offwh[l] = cvt(F(p0+5),  256*256);
    vwh[l]   = cvt(F(p0+7),  256*256);
    f1wh[l]  = cvt(F(p0+9),  (size_t)DFF*256);
    f2wh[l]  = cvt(F(p0+11), (size_t)256*DFF);
    inwh[l]  = cvt(F(p0+19), (size_t)768*256);
    outwh[l] = cvt(F(p0+21), 256*256);
  }
  _Float16* CWh = alloch(DB);   // filled after CW is computed

  float* out = (float*)d_out;
  const long long PL = (long long)BS * NQ * NTXT;     // 48000
  const long long PM = (long long)BS * NQ * MHW;      // 22118400
  const int MS = BS * LSRC;   // 24480 (batch folded into M for shared weights)
  const int MD = BS * LTOT;   // 760

  // ---------- encoder-side prep ----------
  build_src_k<<<g1(SB), 256, 0, stream>>>(x0, x1, x2, x3, SRC);
  props_k<<<g1(LSRC), 256, 0, stream>>>(PU, VAL);
  mask_mem_k<<<g1(SB), 256, 0, stream>>>(SRC, VAL, T1);
  gemm(true,false, T1, encwh, encbias, T2, MS, HID, HID, 0,0,0, 1, stream);
  add_ln_k<<<MS, 32, 0, stream>>>(T2, nullptr, enclng, enclnb, MEM);
  add_ln_k<<<MS, 32, 0, stream>>>(MEM, nullptr, decg, decb, T1);
  gemm(false,false, T1, clswh, nullptr, T2, MS, HID, HID, 0,0,0, 1, stream);     // cls
  gemm(true,false, T2, txth, nullptr, SCR, LSRC, NTXT, HID,
       (long long)LSRC*HID, (long long)NTXT*HID, (long long)LSRC*NTXT, BS, stream);
  rowmax_k<<<g1((long long)BS*LSRC), 256, 0, stream>>>(SCR, SMX, (long long)BS*LSRC);
  gemm(true,true,  MEM, eb1wh, eb1b, T1, MS, HID, HID, 0,0,0, 1, stream);
  gemm(true,true,  T1,  eb2wh, eb2b, T2, MS, HID, HID, 0,0,0, 1, stream);
  gemm(true,false, T2,  eb3wh, eb3b, CU, MS, 4, HID, 0,0,0, 1, stream);
  coords_add_k<<<g1((long long)BS*LSRC*4), 256, 0, stream>>>(CU, PU);
  { dim3 gr((LSRC + 255) / 256, BS); topk_rank_k<<<gr, 256, 0, stream>>>(SMX, TIDX); }
  init_tgt_k<<<g1(DB), 256, 0, stream>>>(MEM, TIDX, lab, pb, txt, OUT);
  init_ref_k<<<g1((long long)BS*LTOT*4), 256, 0, stream>>>(CU, TIDX, REF);
  build_mask_k<<<g1((long long)LTOT*LTOT), 256, 0, stream>>>(MASK);

  // ---------- decoder layers ----------
  for (int l = 0; l < NLAYER; ++l) {
    int p0 = 26 + 22 * l;
    const float *awb=F(p0+0), *ob=F(p0+2), *offb=F(p0+4), *vb=F(p0+6);
    const float *f1b=F(p0+8), *f2b=F(p0+10);
    const float *n1b=F(p0+12), *n1g=F(p0+13), *n2b=F(p0+14), *n2g=F(p0+15);
    const float *n3b=F(p0+16), *n3g=F(p0+17);
    const float *inb=F(p0+18), *outb=F(p0+20);

    // query position embedding
    sine_embed_k<<<g1((long long)BS*LTOT*512), 256, 0, stream>>>(REF, SE);
    gemm(true,true,  SE,  rp1wh, rp1b, DT1, MD, HID, 512, 0,0,0, 1, stream);
    gemm(true,false, DT1, rp2wh, rp2b, QPOS, MD, HID, HID, 0,0,0, 1, stream);
    ew_add_k<<<g1(DB), 256, 0, stream>>>(OUT, QPOS, Q, DB);

    // self-attention
    gemm(true,false, Q,   inwh[l],           inb,       QH, MD, HID, HID, 0,0,0, 1, stream);
    gemm(true,false, Q,   inwh[l] + 256*256, inb + 256, KH, MD, HID, HID, 0,0,0, 1, stream);
    gemm(true,false, OUT, inwh[l] + 512*256, inb + 512, VH, MD, HID, HID, 0,0,0, 1, stream);
    { dim3 gr(LTOT, NHEAD, BS); mha_attn_k<<<gr, 128, 0, stream>>>(QH, KH, VH, MASK, ATT); }
    gemm(true,false, ATT, outwh[l], outb, DT1, MD, HID, HID, 0,0,0, 1, stream);
    add_ln_k<<<MD, 32, 0, stream>>>(OUT, DT1, n2g, n2b, OUT);

    // deformable cross-attention (memory = raw src)
    ew_add_k<<<g1(DB), 256, 0, stream>>>(OUT, QPOS, Q, DB);
    gemm(true,false, SRC, vwh[l],  vb,  T2,  MS, HID, HID, 0,0,0, 1, stream);   // value
    gemm(true,false, Q,   offwh[l], offb, OFFB, MD, HID, HID, 0,0,0, 1, stream);
    gemm(true,false, Q,   awwh[l], awb, AW,  MD, 128, HID, 0,0,0, 1, stream);
    softmax16_k<<<g1((long long)BS*LTOT*NHEAD), 256, 0, stream>>>(AW, (long long)BS*LTOT*NHEAD);
    { dim3 gr(LTOT, NHEAD, BS); deform_attn_k<<<gr, 32, 0, stream>>>(T2, REF, OFFB, AW, ATT); }
    gemm(true,false, ATT, owh[l], ob, DT1, MD, HID, HID, 0,0,0, 1, stream);
    add_ln_k<<<MD, 32, 0, stream>>>(OUT, DT1, n1g, n1b, OUT);

    // FFN
    gemm(true,true,  OUT, f1wh[l], f1b, FFB, MD, DFF, HID, 0,0,0, 1, stream);
    gemm(true,false, FFB, f2wh[l], f2b, DT1, MD, HID, DFF, 0,0,0, 1, stream);
    add_ln_k<<<MD, 32, 0, stream>>>(OUT, DT1, n3g, n3b, OUT);

    // reference refinement (skip on last layer: boxes use refs[8])
    if (l < NLAYER - 1) {
      gemm(true,true,  OUT, bb1wh, bb1b, DT1, MD, HID, HID, 0,0,0, 1, stream);
      gemm(true,true,  DT1, bb2wh, bb2b, DT2, MD, HID, HID, 0,0,0, 1, stream);
      gemm(true,false, DT2, bb3wh, bb3b, BB,  MD, 4,   HID, 0,0,0, 1, stream);
      bbox_sigmoid_k<<<g1((long long)BS*LTOT*4), 256, 0, stream>>>(BB, REF, REF, (long long)BS*LTOT*4);
    }
  }

  // ---------- heads ----------
  add_ln_k<<<MD, 32, 0, stream>>>(OUT,  nullptr, decg, decb, HLST);   // hs[-1]
  add_ln_k<<<MD, 32, 0, stream>>>(HLST, nullptr, decg, decb, DN);     // d
  gemm(false,false, DN, clswh, nullptr, CW, MD, HID, HID, 0,0,0, 1, stream);
  f32_to_f16_k<<<g1((long long)DB), 256, 0, stream>>>(CW, CWh, (long long)DB);
  // pred_logits = cw[:, :300] @ cw[:, 300:]^T   (per batch)
  gemm(true,false, CW, CWh + (long long)NQ*HID, nullptr, out,
       NQ, NTXT, HID, (long long)LTOT*HID, (long long)LTOT*HID, (long long)NQ*NTXT, BS, stream);
  // mask embed MLP on d[:, :300]
  gemm(true,true, DN, me1wh, me1b, ME1, NQ, HID, HID,
       (long long)LTOT*HID, 0, (long long)NQ*HID, BS, stream);
  gemm(true,true,  ME1, me2wh, me2b, ME2, BS*NQ, HID, HID, 0,0,0, 1, stream);
  gemm(true,false, ME2, me3wh, me3b, MEB, BS*NQ, HID, HID, 0,0,0, 1, stream);
  // pred_masks = memb @ mask_features  (per batch, [K=256, N=192*192], f16 B)
  gemm(false,false, MEB, mfh, nullptr, out + PL, NQ, MHW, HID,
       (long long)NQ*HID, (long long)HID*MHW, (long long)NQ*MHW, BS, stream);
  // boxes[-1] = sigmoid(bbox_mlp(hs[-1]) + inverse_sigmoid(refs[8]))
  gemm(true,true,  HLST, bb1wh, bb1b, DT1, MD, HID, HID, 0,0,0, 1, stream);
  gemm(true,true,  DT1,  bb2wh, bb2b, DT2, MD, HID, HID, 0,0,0, 1, stream);
  gemm(true,false, DT2,  bb3wh, bb3b, BB,  MD, 4,   HID, 0,0,0, 1, stream);
  bbox_sigmoid_k<<<g1((long long)BS*LTOT*4), 256, 0, stream>>>(BB, REF, out + PL + PM, (long long)BS*LTOT*4);
}
